// WMInEKFLayer_23622320128496
// MI455X (gfx1250) — matrix operand order repaired and checked by reading, not compiled, not run on hardware
//
#include <hip/hip_runtime.h>
#include <hip/hip_bf16.h>

// ---------------------------------------------------------------------------
// Transformer block for MI455X (gfx1250): bf16 WMMA + double-buffered TDM
// tensor_load_to_lds staging. B=2, T=2048, D=1024, H=16, DH=64, NB=32
// ---------------------------------------------------------------------------

#define BB 2
#define TT 2048
#define DD 1024
#define DH 64
#define HH 16
#define NB 32
#define MROWS (BB * TT)   // 4096

typedef __attribute__((ext_vector_type(16))) __bf16 bf16x16;
typedef __attribute__((ext_vector_type(8)))  float  v8f;
typedef __attribute__((ext_vector_type(4)))  unsigned u32x4;
typedef __attribute__((ext_vector_type(8)))  unsigned u32x8;

union Frag16 {            // 16 bf16 = 8 dwords; A or B operand of 16x16x32 bf16 WMMA
    bf16x16 v;
    unsigned u[8];
};

__device__ __forceinline__ unsigned short f2bf(float f) {
    unsigned u = __float_as_uint(f);
    unsigned r = u + 0x7FFFu + ((u >> 16) & 1u);   // round-to-nearest-even
    return (unsigned short)(r >> 16);
}
__device__ __forceinline__ float bf2f(unsigned short h) {
    return __uint_as_float(((unsigned)h) << 16);
}
__device__ __forceinline__ void ld16B(const unsigned short* p, unsigned* dst) {
    const uint4 t = *(const uint4*)p;
    dst[0] = t.x; dst[1] = t.y; dst[2] = t.z; dst[3] = t.w;
}

// ---------------------------------------------------------------------------
// TDM: 2D bf16 tile DMA, global -> LDS. Descriptor per ISA 08_async_tensor §8.
// 2-group form (VADDR2/3 = NULL) for tensors up to 2D. Issued by one wave.
// ---------------------------------------------------------------------------
__device__ __forceinline__ void tdm_load_2d_bf16(unsigned lds_off, const unsigned short* gp,
                                                 unsigned td0, unsigned td1,
                                                 unsigned tile0, unsigned tile1,
                                                 unsigned stride0) {
    unsigned long long ga = (unsigned long long)(size_t)gp;
    u32x4 g0;
    g0[0] = 1u;                                               // count=1, is_restore=0, gather off
    g0[1] = lds_off;                                          // lds_addr (bytes)
    g0[2] = (unsigned)(ga & 0xFFFFFFFFu);                     // global_addr[31:0]
    g0[3] = (unsigned)((ga >> 32) & 0x1FFFFFFu) | (2u << 30); // global_addr[56:32] | type=2
    u32x8 g1;
    g1[0] = 0x00010000u;                                      // wg_mask=0, data_size=1 (2B)
    g1[1] = (td0 & 0xFFFFu) << 16;                            // atomic_barrier=0, tensor_dim0 lo
    g1[2] = (td0 >> 16) | ((td1 & 0xFFFFu) << 16);            // tensor_dim0 hi, tensor_dim1 lo
    g1[3] = (td1 >> 16) | ((tile0 & 0xFFFFu) << 16);          // tensor_dim1 hi, tile_dim0
    g1[4] = tile1 & 0xFFFFu;                                  // tile_dim1, tile_dim2=0
    g1[5] = stride0;                                          // tensor_dim0_stride[31:0]
    g1[6] = 0u;                                               // stride hi, dim1_stride lo
    g1[7] = 0u;
    asm volatile("tensor_load_to_lds %0, %1" :: "s"(g0), "s"(g1) : "memory");
}

// ---------------------------------------------------------------------------
// fp32 -> bf16 conversion
// ---------------------------------------------------------------------------
__global__ __launch_bounds__(256) void cvt_kernel(const float* __restrict__ in,
                                                  unsigned short* __restrict__ out, int n) {
    int i = blockIdx.x * 256 + threadIdx.x;
    if (i < n) out[i] = f2bf(in[i]);
}

// ---------------------------------------------------------------------------
// LayerNorm over D=1024, one block (256 threads) per row, bf16 output
// ---------------------------------------------------------------------------
__global__ __launch_bounds__(256) void ln_kernel(const float* __restrict__ x,
                                                 const float* __restrict__ g,
                                                 const float* __restrict__ b,
                                                 unsigned short* __restrict__ out) {
    __shared__ float sm[256];
    const int tid = threadIdx.x;
    const float* xr = x + (size_t)blockIdx.x * DD;
    float v[4], s = 0.f;
#pragma unroll
    for (int i = 0; i < 4; i++) { v[i] = xr[tid + i * 256]; s += v[i]; }
    sm[tid] = s; __syncthreads();
    for (int o = 128; o > 0; o >>= 1) { if (tid < o) sm[tid] += sm[tid + o]; __syncthreads(); }
    const float mu = sm[0] * (1.f / DD);
    __syncthreads();
    float vs = 0.f;
#pragma unroll
    for (int i = 0; i < 4; i++) { float d = v[i] - mu; vs += d * d; }
    sm[tid] = vs; __syncthreads();
    for (int o = 128; o > 0; o >>= 1) { if (tid < o) sm[tid] += sm[tid + o]; __syncthreads(); }
    const float rs = rsqrtf(sm[0] * (1.f / DD) + 1e-5f);
    unsigned short* orow = out + (size_t)blockIdx.x * DD;
#pragma unroll
    for (int i = 0; i < 4; i++) {
        int c = tid + i * 256;
        orow[c] = f2bf((v[i] - mu) * rs * g[c] + b[c]);
    }
}

// ---------------------------------------------------------------------------
// temp[b][h][t] = 1 + mean_i sigma[b][t][h][i]
// ---------------------------------------------------------------------------
__global__ __launch_bounds__(256) void temp_kernel(const float* __restrict__ sigma,
                                                   float* __restrict__ temp) {
    int idx = blockIdx.x * 256 + threadIdx.x;          // over B*H*T
    if (idx >= BB * HH * TT) return;
    int t = idx % TT, h = (idx / TT) % HH, b = idx / (TT * HH);
    const float* s = sigma + (((size_t)(b * TT + t)) * HH + h) * NB;
    float acc = 0.f;
#pragma unroll
    for (int i = 0; i < NB; i++) acc += s[i];
    temp[idx] = 1.f + acc * (1.f / NB);
}

// ---------------------------------------------------------------------------
// RoPE in-place on bf16 Q and K (layout [b,t, h*64+d])
// ---------------------------------------------------------------------------
__global__ __launch_bounds__(256) void rope_kernel(unsigned short* __restrict__ Q,
                                                   unsigned short* __restrict__ K,
                                                   const float* __restrict__ cos_a,
                                                   const float* __restrict__ sin_a) {
    int idx = blockIdx.x * 256 + threadIdx.x;          // over B*T*H*NB
    if (idx >= BB * TT * HH * NB) return;
    int i = idx % NB, h = (idx / NB) % HH, t = (idx / (NB * HH)) % TT, b = idx / (NB * HH * TT);
    size_t qoff = ((size_t)(b * TT + t)) * DD + h * DH + 2 * i;
    size_t aoff = (((size_t)(b * HH + h)) * TT + t) * NB + i;
    float c = cos_a[aoff], s = sin_a[aoff];
    float q1 = bf2f(Q[qoff]), q2 = bf2f(Q[qoff + 1]);
    Q[qoff]     = f2bf(q1 * c - q2 * s);
    Q[qoff + 1] = f2bf(q1 * s + q2 * c);
    float k1 = bf2f(K[qoff]), k2 = bf2f(K[qoff + 1]);
    K[qoff]     = f2bf(k1 * c - k2 * s);
    K[qoff + 1] = f2bf(k1 * s + k2 * c);
}

// ---------------------------------------------------------------------------
// bf16 GEMM: C[M,N] = act(A[M,K] @ B[K,N] + bias) (+ resid). 128x128 block,
// 8 waves of 32x64, v_wmma_f32_16x16x32_bf16. Double-buffered staging:
// A tile via TDM DMA (overlaps compute), B tile via global loads held in
// registers across the compute phase, then scattered transposed into LDS.
// act: 0=none, 1=exact GELU
// ---------------------------------------------------------------------------
__global__ __launch_bounds__(256) void gemm_bf16(const unsigned short* __restrict__ A,
                                                 const unsigned short* __restrict__ Bw,
                                                 const float* __restrict__ bias,
                                                 const float* __restrict__ resid,
                                                 float* __restrict__ outF,
                                                 unsigned short* __restrict__ outB,
                                                 int M, int N, int K, int act) {
    __shared__ unsigned short As[2][128 * 32];   // [m][k], row-major (TDM-filled)
    __shared__ unsigned short Bt[2][128 * 32];   // [n][k]  (k-contiguous per column)
    const int tid = threadIdx.x, lane = tid & 31, wid = tid >> 5;
    const int half = lane >> 4, lm = lane & 15;
    const int m0 = blockIdx.y * 128, n0 = blockIdx.x * 128;
    const int wm = (wid & 3) * 32, wn = (wid >> 2) * 64;

    v8f acc[2][4];
#pragma unroll
    for (int i = 0; i < 2; i++)
#pragma unroll
        for (int j = 0; j < 4; j++) acc[i][j] = (v8f){};

    uint4 breg[2];
    // issue stage: TDM for A tile, global loads for B tile into registers
    auto stage_issue = [&](int k0, int buf) {
        if (wid == 0)
            tdm_load_2d_bf16((unsigned)(size_t)(void*)&As[buf][0],
                             A + (size_t)m0 * K + k0,
                             /*td0*/32, /*td1*/128, /*tile0*/32, /*tile1*/128,
                             /*stride0*/(unsigned)K);
#pragma unroll
        for (int i = 0; i < 2; i++) {
            int idx = tid * 2 + i;
            int kr = idx >> 4, nq = idx & 15;
            const unsigned short* gp = Bw + (size_t)(k0 + kr) * N + n0 + nq * 8;
            if (k0 + 32 < K) __builtin_prefetch(gp + (size_t)32 * N, 0, 3);
            breg[i] = *(const uint4*)gp;
        }
    };
    // commit stage: scatter B registers transposed into Bt[buf]
    auto stage_commit = [&](int buf) {
#pragma unroll
        for (int i = 0; i < 2; i++) {
            int idx = tid * 2 + i;
            int kr = idx >> 4, nq = idx & 15;
            const unsigned short* ds = (const unsigned short*)&breg[i];
#pragma unroll
            for (int j = 0; j < 8; j++) Bt[buf][(nq * 8 + j) * 32 + kr] = ds[j];
        }
    };
    // compute stage: 8 WMMAs on tile in buffer `buf`
    auto compute_tile = [&](int buf) {
        Frag16 af[2], bfr[4];
#pragma unroll
        for (int mi = 0; mi < 2; mi++) {               // A 16x32 per-lane layout
            const unsigned short* p = &As[buf][(wm + mi * 16 + lm) * 32];
            ld16B(p + half * 8, &af[mi].u[0]);         // K = half*8 .. +7
            ld16B(p + 16 + half * 8, &af[mi].u[4]);    // K = 16+half*8 .. +7
        }
#pragma unroll
        for (int nj = 0; nj < 4; nj++) {               // B 32x16 per-lane layout
            const unsigned short* p = &Bt[buf][(wn + nj * 16 + lm) * 32 + half * 8];
            ld16B(p, &bfr[nj].u[0]);                   // K = half*8 .. +7
            ld16B(p + 16, &bfr[nj].u[4]);              // K = 16+half*8 .. +7
        }
#pragma unroll
        for (int mi = 0; mi < 2; mi++)
#pragma unroll
            for (int nj = 0; nj < 4; nj++)
                acc[mi][nj] = __builtin_amdgcn_wmma_f32_16x16x32_bf16(
                    false, af[mi].v, false, bfr[nj].v, (short)0, acc[mi][nj], false, false);
    };

    // software pipeline: fetch tile i+1 while computing tile i
    stage_issue(0, 0);
    stage_commit(0);
    if (wid == 0) __builtin_amdgcn_s_wait_tensorcnt(0);
    __syncthreads();
    int buf = 0;
    for (int k0 = 32; k0 < K; k0 += 32) {
        stage_issue(k0, buf ^ 1);      // TDM + global loads in flight...
        compute_tile(buf);             // ...overlapped with WMMA compute
        stage_commit(buf ^ 1);
        if (wid == 0) __builtin_amdgcn_s_wait_tensorcnt(0);
        __syncthreads();
        buf ^= 1;
    }
    compute_tile(buf);

    // epilogue: C layout => row = base + 8*half + r, col = base + lm
#pragma unroll
    for (int mi = 0; mi < 2; mi++)
#pragma unroll
        for (int nj = 0; nj < 4; nj++)
#pragma unroll
            for (int r = 0; r < 8; r++) {
                int gm = m0 + wm + mi * 16 + half * 8 + r;
                int gn = n0 + wn + nj * 16 + lm;
                float v = acc[mi][nj][r] + bias[gn];
                if (act == 1) v = 0.5f * v * (1.0f + erff(v * 0.70710678118654752f));
                if (resid) v += resid[(size_t)gm * N + gn];
                size_t o = (size_t)gm * N + gn;
                if (outF) outF[o] = v;
                if (outB) outB[o] = f2bf(v);
            }
}

// ---------------------------------------------------------------------------
// Flash attention, causal, per-row temperature. One block = (b,h) x 128 queries,
// 8 waves x 16 query rows. Key tiles of 32, double-buffered: K tile via TDM,
// V tile via global loads held in registers then transposed into LDS.
// bf16 WMMA for QK^T and P*V.
// ---------------------------------------------------------------------------
__global__ __launch_bounds__(256) void attn_kernel(const unsigned short* __restrict__ Qb,
                                                   const unsigned short* __restrict__ Kb,
                                                   const unsigned short* __restrict__ Vb,
                                                   const float* __restrict__ temp,
                                                   unsigned short* __restrict__ Ob) {
    __shared__ unsigned short Ks[2][32 * 64];    // [s][d] row-major (TDM-filled)
    __shared__ unsigned short Vt[2][64 * 32];    // [d][s] transposed (B operand of P*V)
    __shared__ unsigned short Ps[8][16 * 32];    // per-wave P staging (C->A layout)

    const int tid = threadIdx.x, lane = tid & 31, wid = tid >> 5;
    const int half = lane >> 4, lm = lane & 15;
    const int bh = blockIdx.y, b = bh >> 4, h = bh & 15;
    const int q0 = blockIdx.x * 128;
    const int qw = q0 + wid * 16;

    // load Q fragments (16 rows x 64 d = two 16x32 A-fragments), fixed per wave
    Frag16 aq[2];
    {
        const unsigned short* qp = Qb + ((size_t)(b * TT + qw + lm)) * DD + h * DH;
#pragma unroll
        for (int c = 0; c < 2; c++) {
            ld16B(qp + c * 32 + half * 8, &aq[c].u[0]);
            ld16B(qp + c * 32 + 16 + half * 8, &aq[c].u[4]);
        }
    }
    float srow[8], mi[8], li[8];
#pragma unroll
    for (int r = 0; r < 8; r++) {
        int t = qw + half * 8 + r;
        srow[r] = 1.f / (8.f * temp[(size_t)bh * TT + t]);
        mi[r] = -1e30f; li[r] = 0.f;
    }
    v8f accO[4];
#pragma unroll
    for (int j = 0; j < 4; j++) accO[j] = (v8f){};

    uint4 vreg;
    auto stage_issue = [&](int s0, int buf) {
        if (wid == 0)
            tdm_load_2d_bf16((unsigned)(size_t)(void*)&Ks[buf][0],
                             Kb + ((size_t)(b * TT + s0)) * DD + h * DH,
                             /*td0*/64, /*td1*/32, /*tile0*/64, /*tile1*/32,
                             /*stride0*/DD);
        int s = tid >> 3, dq = tid & 7;
        vreg = *(const uint4*)(Vb + ((size_t)(b * TT + s0 + s)) * DD + h * DH + dq * 8);
    };
    auto stage_commit = [&](int buf) {
        int s = tid >> 3, dq = tid & 7;
        const unsigned short* vs = (const unsigned short*)&vreg;
#pragma unroll
        for (int j = 0; j < 8; j++) Vt[buf][(dq * 8 + j) * 32 + s] = vs[j];
    };

    auto compute_tile = [&](int s0, int buf) {
        // S = Q K^T for two 16-key subtiles, scaled + masked
        float sv[2][8];
#pragma unroll
        for (int ns = 0; ns < 2; ns++) {
            v8f sacc = (v8f){};
#pragma unroll
            for (int c = 0; c < 2; c++) {
                Frag16 bk;
                const unsigned short* p = &Ks[buf][(ns * 16 + lm) * 64 + c * 32 + half * 8];
                ld16B(p, &bk.u[0]);
                ld16B(p + 16, &bk.u[4]);
                sacc = __builtin_amdgcn_wmma_f32_16x16x32_bf16(
                    false, aq[c].v, false, bk.v, (short)0, sacc, false, false);
            }
#pragma unroll
            for (int r = 0; r < 8; r++) {
                int t = qw + half * 8 + r;
                int s = s0 + ns * 16 + lm;
                float v = sacc[r] * srow[r];
                sv[ns][r] = (s > t) ? -1e30f : v;
            }
        }
        // online softmax (row = 16 lanes within half-wave)
        float alpha[8];
#pragma unroll
        for (int r = 0; r < 8; r++) {
            float rm = fmaxf(sv[0][r], sv[1][r]);
#pragma unroll
            for (int o = 1; o < 16; o <<= 1) rm = fmaxf(rm, __shfl_xor(rm, o, 32));
            float mn = fmaxf(mi[r], rm);
            alpha[r] = __expf(mi[r] - mn);
            float p0 = __expf(sv[0][r] - mn);
            float p1 = __expf(sv[1][r] - mn);
            float rs = p0 + p1;
#pragma unroll
            for (int o = 1; o < 16; o <<= 1) rs += __shfl_xor(rs, o, 32);
            li[r] = li[r] * alpha[r] + rs;
            mi[r] = mn;
            sv[0][r] = p0; sv[1][r] = p1;
        }
#pragma unroll
        for (int j = 0; j < 4; j++)
#pragma unroll
            for (int r = 0; r < 8; r++) accO[j][r] *= alpha[r];

        // P: C-layout -> LDS -> A-layout fragment (wave-private region)
#pragma unroll
        for (int ns = 0; ns < 2; ns++)
#pragma unroll
            for (int r = 0; r < 8; r++)
                Ps[wid][(half * 8 + r) * 32 + ns * 16 + lm] = f2bf(sv[ns][r]);
        Frag16 pa;
        {
            const unsigned short* pp = &Ps[wid][lm * 32];
            ld16B(pp + half * 8, &pa.u[0]);
            ld16B(pp + 16 + half * 8, &pa.u[4]);
        }
        // O += P * V   (V as 32x64 => 4 column subtiles)
#pragma unroll
        for (int j = 0; j < 4; j++) {
            Frag16 vb;
            const unsigned short* p = &Vt[buf][(j * 16 + lm) * 32 + half * 8];
            ld16B(p, &vb.u[0]);
            ld16B(p + 16, &vb.u[4]);
            accO[j] = __builtin_amdgcn_wmma_f32_16x16x32_bf16(
                false, pa.v, false, vb.v, (short)0, accO[j], false, false);
        }
    };

    const int send = q0 + 128;  // causal: only keys <= q0+127 contribute
    // software pipeline over key tiles
    stage_issue(0, 0);
    stage_commit(0);
    if (wid == 0) __builtin_amdgcn_s_wait_tensorcnt(0);
    __syncthreads();
    int buf = 0;
    for (int s0 = 32; s0 < send; s0 += 32) {
        stage_issue(s0, buf ^ 1);      // next K tile DMA + V loads in flight...
        compute_tile(s0 - 32, buf);    // ...overlapped with S/P/O compute
        stage_commit(buf ^ 1);
        if (wid == 0) __builtin_amdgcn_s_wait_tensorcnt(0);
        __syncthreads();
        buf ^= 1;
    }
    compute_tile(send - 32, buf);

    // finalize: O /= l, store bf16 at [b, t, h*64+d]
#pragma unroll
    for (int r = 0; r < 8; r++) {
        float inv = 1.f / li[r];
        int t = qw + half * 8 + r;
#pragma unroll
        for (int j = 0; j < 4; j++) {
            int d = j * 16 + lm;
            Ob[((size_t)(b * TT + t)) * DD + h * DH + d] = f2bf(accO[j][r] * inv);
        }
    }
}

// ---------------------------------------------------------------------------
// Launcher
// ---------------------------------------------------------------------------
extern "C" void kernel_launch(void* const* d_in, const int* in_sizes, int n_in,
                              void* d_out, int out_size, void* d_ws, size_t ws_size,
                              hipStream_t stream) {
    const float* x     = (const float*)d_in[0];
    const float* cos_a = (const float*)d_in[1];
    const float* sin_a = (const float*)d_in[2];
    const float* sigma = (const float*)d_in[3];
    // d_in[4] = causal_mask (implicit in attn kernel)
    const float* ln1_g = (const float*)d_in[5];
    const float* ln1_b = (const float*)d_in[6];
    const float* wq = (const float*)d_in[7];  const float* bq = (const float*)d_in[8];
    const float* wk = (const float*)d_in[9];  const float* bk = (const float*)d_in[10];
    const float* wv = (const float*)d_in[11]; const float* bv = (const float*)d_in[12];
    const float* wo = (const float*)d_in[13]; const float* bo = (const float*)d_in[14];
    const float* ln2_g = (const float*)d_in[15];
    const float* ln2_b = (const float*)d_in[16];
    const float* w1 = (const float*)d_in[17]; const float* b1 = (const float*)d_in[18];
    const float* w2 = (const float*)d_in[19]; const float* b2 = (const float*)d_in[20];
    float* out = (float*)d_out;

    // bump allocator over d_ws
    char* wp = (char*)d_ws;
    auto alloc = [&](size_t bytes) -> void* {
        void* p = (void*)wp;
        wp += (bytes + 255) & ~(size_t)255;
        return p;
    };
    const size_t DxD = (size_t)DD * DD;           // 1M
    const size_t MxD = (size_t)MROWS * DD;        // 4M
    unsigned short* wq_bf = (unsigned short*)alloc(DxD * 2);
    unsigned short* wk_bf = (unsigned short*)alloc(DxD * 2);
    unsigned short* wv_bf = (unsigned short*)alloc(DxD * 2);
    unsigned short* wo_bf = (unsigned short*)alloc(DxD * 2);
    unsigned short* w1_bf = (unsigned short*)alloc(DxD * 4 * 2);
    unsigned short* w2_bf = (unsigned short*)alloc(DxD * 4 * 2);
    unsigned short* h_bf  = (unsigned short*)alloc(MxD * 2);
    unsigned short* Qbf   = (unsigned short*)alloc(MxD * 2);
    unsigned short* Kbf   = (unsigned short*)alloc(MxD * 2);
    unsigned short* Vbf   = (unsigned short*)alloc(MxD * 2);
    unsigned short* Obf   = (unsigned short*)alloc(MxD * 2);
    float*          tempb = (float*)alloc((size_t)BB * HH * TT * 4);
    float*          x2    = (float*)alloc(MxD * 4);
    unsigned short* h2bf  = (unsigned short*)alloc(MxD * 2);
    unsigned short* gbf   = (unsigned short*)alloc(MxD * 4 * 2);

    // 1) weights -> bf16
    cvt_kernel<<<(int)(DxD / 256), 256, 0, stream>>>(wq, wq_bf, (int)DxD);
    cvt_kernel<<<(int)(DxD / 256), 256, 0, stream>>>(wk, wk_bf, (int)DxD);
    cvt_kernel<<<(int)(DxD / 256), 256, 0, stream>>>(wv, wv_bf, (int)DxD);
    cvt_kernel<<<(int)(DxD / 256), 256, 0, stream>>>(wo, wo_bf, (int)DxD);
    cvt_kernel<<<(int)(DxD * 4 / 256), 256, 0, stream>>>(w1, w1_bf, (int)(DxD * 4));
    cvt_kernel<<<(int)(DxD * 4 / 256), 256, 0, stream>>>(w2, w2_bf, (int)(DxD * 4));

    // 2) LN1 -> h (bf16)
    ln_kernel<<<MROWS, 256, 0, stream>>>(x, ln1_g, ln1_b, h_bf);

    // 3) Q/K/V projections (WMMA GEMM, bf16 out)
    dim3 g_qkv(DD / 128, MROWS / 128);
    gemm_bf16<<<g_qkv, 256, 0, stream>>>(h_bf, wq_bf, bq, nullptr, nullptr, Qbf, MROWS, DD, DD, 0);
    gemm_bf16<<<g_qkv, 256, 0, stream>>>(h_bf, wk_bf, bk, nullptr, nullptr, Kbf, MROWS, DD, DD, 0);
    gemm_bf16<<<g_qkv, 256, 0, stream>>>(h_bf, wv_bf, bv, nullptr, nullptr, Vbf, MROWS, DD, DD, 0);

    // 4) temperature + RoPE
    temp_kernel<<<(BB * HH * TT) / 256, 256, 0, stream>>>(sigma, tempb);
    rope_kernel<<<(BB * TT * HH * NB) / 256, 256, 0, stream>>>(Qbf, Kbf, cos_a, sin_a);

    // 5) flash attention
    dim3 g_att(TT / 128, BB * HH);
    attn_kernel<<<g_att, 256, 0, stream>>>(Qbf, Kbf, Vbf, tempb, Obf);

    // 6) out projection + residual -> x2 (fp32)
    gemm_bf16<<<g_qkv, 256, 0, stream>>>(Obf, wo_bf, bo, x, x2, nullptr, MROWS, DD, DD, 0);

    // 7) LN2 -> h2 (bf16)
    ln_kernel<<<MROWS, 256, 0, stream>>>(x2, ln2_g, ln2_b, h2bf);

    // 8) FFN1 with exact GELU -> g (bf16)
    dim3 g_f1(DD * 4 / 128, MROWS / 128);
    gemm_bf16<<<g_f1, 256, 0, stream>>>(h2bf, w1_bf, b1, nullptr, nullptr, gbf, MROWS, DD * 4, DD, 1);

    // 9) FFN2 + residual -> out (fp32)
    gemm_bf16<<<g_qkv, 256, 0, stream>>>(gbf, w2_bf, b2, x2, out, nullptr, MROWS, DD, DD * 4, 0);
}
